// PointDecoder_74577812128593
// MI455X (gfx1250) — matrix operand order repaired and checked by reading, not compiled, not run on hardware
//
#include <hip/hip_runtime.h>
#include <hip/hip_bf16.h>
#include <math.h>

// Problem constants (fixed by the reference).
#define BATCH 2048
#define HID   1024
#define NCLS  2
#define TSTEP 128
#define MT    2          // M tiles (of 16 rows) per wave

typedef __attribute__((ext_vector_type(16))) __bf16 v16bf;
typedef __attribute__((ext_vector_type(8)))  __bf16 v8bf;
typedef __attribute__((ext_vector_type(8)))  float  v8f;

static __device__ __forceinline__ float sigmoidf_(float x) {
    return 1.0f / (1.0f + __expf(-x));
}

static __device__ __forceinline__ v16bf cat16(v8bf lo, v8bf hi) {
    return __builtin_shufflevector(lo, hi, 0, 1, 2, 3, 4, 5, 6, 7,
                                           8, 9, 10, 11, 12, 13, 14, 15);
}

// A fragment: 16x32 bf16 tile of a row-major [.,HID] matrix.
// Lane l: row = base + (l&15); K offset (l>>4)*16; 16 contiguous K values.
static __device__ __forceinline__ v16bf load_a(const __bf16* __restrict__ p) {
    v8bf lo = *reinterpret_cast<const v8bf*>(p);
    v8bf hi = *reinterpret_cast<const v8bf*>(p + 8);
    return cat16(lo, hi);
}

// B fragment: 32x16 bf16 tile representing W^T, loaded from row-major W [N,K].
// Lane l: col n = base + (l&15); K offset (l>>4)*16; 16 contiguous K values.
static __device__ __forceinline__ v16bf load_b(const __bf16* __restrict__ p) {
    v8bf lo = *reinterpret_cast<const v8bf*>(p);
    v8bf hi = *reinterpret_cast<const v8bf*>(p + 8);
    return cat16(lo, hi);
}

static __device__ __forceinline__ v8f wmma_bf16(v16bf a, v16bf b, v8f c) {
    return __builtin_amdgcn_wmma_f32_16x16x32_bf16(
        false, a, false, b, (short)0, c, false, false);
}

static __device__ __forceinline__ void load_frags(v16bf (&a)[MT], v16bf (&b)[4],
                                                  const __bf16* __restrict__ aPtr,
                                                  const __bf16* __restrict__ bPtr) {
#pragma unroll
    for (int mt = 0; mt < MT; ++mt) a[mt] = load_a(aPtr + (size_t)mt * 16 * HID);
#pragma unroll
    for (int g = 0; g < 4; ++g) b[g] = load_b(bPtr + (size_t)g * HID * HID);
}

static __device__ __forceinline__ void do_mmas(v8f (&acc)[MT][4],
                                               v16bf (&a)[MT], v16bf (&b)[4]) {
#pragma unroll
    for (int g = 0; g < 4; ++g)
#pragma unroll
        for (int mt = 0; mt < MT; ++mt)
            acc[mt][g] = wmma_bf16(a[mt], b[g], acc[mt][g]);
}

// One full K sweep (K = HID) of gates += A @ W^T for MT M-tiles x 4 gates.
// Manually unrolled x2 with alternating fragment buffers: the ping-pong is
// baked into the loop body so the register allocator needs NO cross-buffer
// copies on the back edge (previous version burned ~48 v_mov + hazard NOPs
// per iteration on this). Loads for the next half-step issue before the
// current half-step's WMMAs -> partial s_wait_loadcnt overlap with L2.
// NOTE: the final prefetch reads 32 elements past the K range; it lands in
// the next (padded) workspace allocation and its value is never consumed.
static __device__ __forceinline__ void kloop(v8f (&acc)[MT][4],
                                             const __bf16* __restrict__ aPtr,
                                             const __bf16* __restrict__ bPtr)
{
    v16bf a0[MT], b0[4], a1[MT], b1[4];
    load_frags(a0, b0, aPtr, bPtr);                       // k = 0
    for (int k = 0; k < HID; k += 64) {
        load_frags(a1, b1, aPtr + 32, bPtr + 32);         // k + 32
        do_mmas(acc, a0, b0);                             // consume k
        load_frags(a0, b0, aPtr + 64, bPtr + 64);         // k + 64 (dead on last iter)
        do_mmas(acc, a1, b1);                             // consume k + 32
        aPtr += 64;
        bPtr += 64;
    }
}

// ---------------- setup kernels ----------------

__global__ void f32_to_bf16_kernel(const float* __restrict__ in,
                                   __bf16* __restrict__ out, int n) {
    int i = blockIdx.x * blockDim.x + threadIdx.x;
    if (i < n) out[i] = (__bf16)in[i];
}

__global__ void copy_f32_kernel(const float* __restrict__ in,
                                float* __restrict__ out, int n) {
    int i = blockIdx.x * blockDim.x + threadIdx.x;
    if (i < n) out[i] = in[i];
}

__global__ void bias_combine_kernel(const float* __restrict__ a,
                                    const float* __restrict__ b,
                                    float* __restrict__ out, int n) {
    int i = blockIdx.x * blockDim.x + threadIdx.x;
    if (i < n) out[i] = a[i] + b[i];
}

__global__ void init_cls_kernel(int* __restrict__ cls, int n) {
    int i = blockIdx.x * blockDim.x + threadIdx.x;
    if (i < n) cls[i] = -1;   // -1 => x_t is the all-zero vector (t==0)
}

// ---------------- layer 0 cell: gates = h0 @ Whh0^T + onehot@Wih0^T + b ----------------
__global__ __launch_bounds__(256)
void lstm_cell0_kernel(const __bf16* __restrict__ h_in,   // [BATCH,HID]
                       __bf16* __restrict__ h_out,        // [BATCH,HID]
                       float* __restrict__ c,             // [BATCH,HID] in/out
                       const __bf16* __restrict__ Whh,    // [4H,H] bf16
                       const float* __restrict__ Wih,     // [4H,NCLS] f32
                       const float* __restrict__ bias,    // [4H] (b_ih+b_hh)
                       const int* __restrict__ cls)       // [BATCH]
{
    const int wid  = (blockIdx.x * blockDim.x + threadIdx.x) >> 5;
    const int lane = threadIdx.x & 31;
    const int mBase = (wid >> 6) * (16 * MT);   // 128/MT M groups
    const int nBase = (wid & 63) << 4;          // 64 N tiles per gate
    const int half  = lane >> 4;

    v8f acc[MT][4];
    const v8f zero = {0.f, 0.f, 0.f, 0.f, 0.f, 0.f, 0.f, 0.f};
#pragma unroll
    for (int mt = 0; mt < MT; ++mt)
#pragma unroll
        for (int g = 0; g < 4; ++g) acc[mt][g] = zero;

    kloop(acc,
          h_in + (size_t)(mBase + (lane & 15)) * HID + half * 16,
          Whh  + (size_t)(nBase + (lane & 15)) * HID + half * 16);

    const int n = nBase + (lane & 15);
    const float bi = bias[0 * HID + n], bf = bias[1 * HID + n];
    const float bg = bias[2 * HID + n], bo = bias[3 * HID + n];

#pragma unroll
    for (int mt = 0; mt < MT; ++mt) {
#pragma unroll
        for (int r = 0; r < 8; ++r) {
            const int m = mBase + mt * 16 + half * 8 + r;  // D layout
            float gi = acc[mt][0][r] + bi;
            float gf = acc[mt][1][r] + bf;
            float gg = acc[mt][2][r] + bg;
            float go = acc[mt][3][r] + bo;
            const int cl = cls[m];
            if (cl >= 0) {                                  // one-hot input term
                gi += Wih[(0 * HID + n) * NCLS + cl];
                gf += Wih[(1 * HID + n) * NCLS + cl];
                gg += Wih[(2 * HID + n) * NCLS + cl];
                go += Wih[(3 * HID + n) * NCLS + cl];
            }
            const size_t idx = (size_t)m * HID + n;
            const float cNew = sigmoidf_(gf) * c[idx] + sigmoidf_(gi) * tanhf(gg);
            c[idx] = cNew;
            h_out[idx] = (__bf16)(sigmoidf_(go) * tanhf(cNew));
        }
    }
}

// ---------------- layer 1 cell: gates = h0new @ Wih1^T + h1 @ Whh1^T + b ----------------
__global__ __launch_bounds__(256)
void lstm_cell1_kernel(const __bf16* __restrict__ x_in,   // h0_new [BATCH,HID]
                       const __bf16* __restrict__ h_in,   // h1     [BATCH,HID]
                       __bf16* __restrict__ h_out,        // h1_new
                       float* __restrict__ c,             // c1 in/out
                       const __bf16* __restrict__ Wih,    // [4H,H]
                       const __bf16* __restrict__ Whh,    // [4H,H]
                       const float* __restrict__ bias)    // [4H]
{
    const int wid  = (blockIdx.x * blockDim.x + threadIdx.x) >> 5;
    const int lane = threadIdx.x & 31;
    const int mBase = (wid >> 6) * (16 * MT);
    const int nBase = (wid & 63) << 4;
    const int half  = lane >> 4;

    v8f acc[MT][4];
    const v8f zero = {0.f, 0.f, 0.f, 0.f, 0.f, 0.f, 0.f, 0.f};
#pragma unroll
    for (int mt = 0; mt < MT; ++mt)
#pragma unroll
        for (int g = 0; g < 4; ++g) acc[mt][g] = zero;

    const size_t aOff = (size_t)(mBase + (lane & 15)) * HID + half * 16;
    const size_t bOff = (size_t)(nBase + (lane & 15)) * HID + half * 16;

    kloop(acc, x_in + aOff, Wih + bOff);   // x @ Wih^T
    kloop(acc, h_in + aOff, Whh + bOff);   // h @ Whh^T

    const int n = nBase + (lane & 15);
    const float bi = bias[0 * HID + n], bf = bias[1 * HID + n];
    const float bg = bias[2 * HID + n], bo = bias[3 * HID + n];

#pragma unroll
    for (int mt = 0; mt < MT; ++mt) {
#pragma unroll
        for (int r = 0; r < 8; ++r) {
            const int m = mBase + mt * 16 + half * 8 + r;
            const float gi = acc[mt][0][r] + bi;
            const float gf = acc[mt][1][r] + bf;
            const float gg = acc[mt][2][r] + bg;
            const float go = acc[mt][3][r] + bo;
            const size_t idx = (size_t)m * HID + n;
            const float cNew = sigmoidf_(gf) * c[idx] + sigmoidf_(gi) * tanhf(gg);
            c[idx] = cNew;
            h_out[idx] = (__bf16)(sigmoidf_(go) * tanhf(cNew));
        }
    }
}

// ---------------- FC head + argmax (one wave per batch row) ----------------
__global__ __launch_bounds__(256)
void fc_argmax_kernel(const __bf16* __restrict__ h1,     // [BATCH,HID]
                      const float* __restrict__ fc_w,    // [NCLS,HID]
                      const float* __restrict__ fc_b,    // [NCLS]
                      float* __restrict__ out,           // [BATCH,TSTEP,NCLS]
                      int* __restrict__ cls, int t)
{
    const int m    = (blockIdx.x * blockDim.x + threadIdx.x) >> 5;
    const int lane = threadIdx.x & 31;
    float s0 = 0.f, s1 = 0.f;
    const __bf16* hp = h1 + (size_t)m * HID;
    for (int k = lane; k < HID; k += 32) {
        const float hv = (float)hp[k];
        s0 += hv * fc_w[k];
        s1 += hv * fc_w[HID + k];
    }
#pragma unroll
    for (int off = 16; off > 0; off >>= 1) {
        s0 += __shfl_xor(s0, off, 32);
        s1 += __shfl_xor(s1, off, 32);
    }
    if (lane == 0) {
        s0 += fc_b[0];
        s1 += fc_b[1];
        float* o = out + (size_t)m * TSTEP * NCLS + (size_t)t * NCLS;
        o[0] = s0;
        o[1] = s1;
        cls[m] = (s1 > s0) ? 1 : 0;   // argmax, first-max tie break
    }
}

// ---------------- host-side orchestration ----------------

extern "C" void kernel_launch(void* const* d_in, const int* in_sizes, int n_in,
                              void* d_out, int out_size, void* d_ws, size_t ws_size,
                              hipStream_t stream) {
    (void)in_sizes; (void)n_in; (void)out_size; (void)ws_size;

    const float* h_init = (const float*)d_in[0];   // [2,B,H]
    const float* c_init = (const float*)d_in[1];   // [2,B,H]
    const float* W_ih0  = (const float*)d_in[3];   // [4H,2]
    const float* W_hh0  = (const float*)d_in[4];   // [4H,H]
    const float* b_ih0  = (const float*)d_in[5];
    const float* b_hh0  = (const float*)d_in[6];
    const float* W_ih1  = (const float*)d_in[7];
    const float* W_hh1  = (const float*)d_in[8];
    const float* b_ih1  = (const float*)d_in[9];
    const float* b_hh1  = (const float*)d_in[10];
    const float* fc_w   = (const float*)d_in[11];  // [2,H]
    const float* fc_b   = (const float*)d_in[12];
    float* out = (float*)d_out;

    const size_t BH = (size_t)BATCH * HID;       // 2M elements
    const size_t WN = (size_t)4 * HID * HID;     // 4M elements

    char* p = (char*)d_ws;
    auto alloc = [&](size_t bytes) -> char* {
        char* r = p;
        p += (bytes + 255) & ~(size_t)255;
        return r;
    };
    __bf16* h0b[2] = {(__bf16*)alloc(BH * 2), (__bf16*)alloc(BH * 2)};
    __bf16* h1b[2] = {(__bf16*)alloc(BH * 2), (__bf16*)alloc(BH * 2)};
    float*  c0     = (float*)alloc(BH * 4);
    float*  c1     = (float*)alloc(BH * 4);
    __bf16* Whh0b  = (__bf16*)alloc(WN * 2);
    __bf16* Wih1b  = (__bf16*)alloc(WN * 2);
    __bf16* Whh1b  = (__bf16*)alloc(WN * 2);
    float*  bias0  = (float*)alloc(4 * HID * 4);
    float*  bias1  = (float*)alloc(4 * HID * 4);
    int*    cls    = (int*)alloc(BATCH * 4);
    (void)alloc(256);   // guard region for the tail prefetch overrun

    // --- setup: weight/bias prep + state init (captured every call; deterministic) ---
    const int cvtBlk = 256;
    f32_to_bf16_kernel<<<(int)((WN + cvtBlk - 1) / cvtBlk), cvtBlk, 0, stream>>>(W_hh0, Whh0b, (int)WN);
    f32_to_bf16_kernel<<<(int)((WN + cvtBlk - 1) / cvtBlk), cvtBlk, 0, stream>>>(W_ih1, Wih1b, (int)WN);
    f32_to_bf16_kernel<<<(int)((WN + cvtBlk - 1) / cvtBlk), cvtBlk, 0, stream>>>(W_hh1, Whh1b, (int)WN);
    f32_to_bf16_kernel<<<(int)((BH + cvtBlk - 1) / cvtBlk), cvtBlk, 0, stream>>>(h_init, h0b[0], (int)BH);
    f32_to_bf16_kernel<<<(int)((BH + cvtBlk - 1) / cvtBlk), cvtBlk, 0, stream>>>(h_init + BH, h1b[0], (int)BH);
    copy_f32_kernel<<<(int)((BH + cvtBlk - 1) / cvtBlk), cvtBlk, 0, stream>>>(c_init, c0, (int)BH);
    copy_f32_kernel<<<(int)((BH + cvtBlk - 1) / cvtBlk), cvtBlk, 0, stream>>>(c_init + BH, c1, (int)BH);
    bias_combine_kernel<<<(4 * HID) / cvtBlk, cvtBlk, 0, stream>>>(b_ih0, b_hh0, bias0, 4 * HID);
    bias_combine_kernel<<<(4 * HID) / cvtBlk, cvtBlk, 0, stream>>>(b_ih1, b_hh1, bias1, 4 * HID);
    init_cls_kernel<<<BATCH / cvtBlk, cvtBlk, 0, stream>>>(cls, BATCH);

    // --- sequential rollout: 3 kernels per step, ping-pong h buffers ---
    const int cellBlocks = ((128 / MT) * 64) / 8;   // waves / (8 waves per block)
    const int fcBlocks   = (BATCH * 32) / 256;
    for (int t = 0; t < TSTEP; ++t) {
        const int rb = t & 1, wb = rb ^ 1;
        lstm_cell0_kernel<<<cellBlocks, 256, 0, stream>>>(
            h0b[rb], h0b[wb], c0, Whh0b, W_ih0, bias0, cls);
        lstm_cell1_kernel<<<cellBlocks, 256, 0, stream>>>(
            h0b[wb], h1b[rb], h1b[wb], c1, Wih1b, Whh1b, bias1);
        fc_argmax_kernel<<<fcBlocks, 256, 0, stream>>>(
            h1b[wb], fc_w, fc_b, out, cls, t);
    }
}
